// Model_50362786513557
// MI455X (gfx1250) — matrix operand order repaired and checked by reading, not compiled, not run on hardware
//
#include <hip/hip_runtime.h>
#include <hip/hip_bf16.h>
#include <math.h>

// ---------------- problem constants ----------------
#define BB   64
#define LL   48
#define VV   50000   // vocab size (row VV of embp is the zero row)
#define EE   300     // embedding / channel dim
#define F0C  300     // fc0 output dim
#define NPAD 320     // padded N (20 tiles of 16 -> constant per-wave tile counts)
#define EPAD 320     // padded K for joint GEMM (multiple of 32), row stride everywhere
#define KC   960     // padded K for conv GEMM: 3 taps * 320
#define NEGV -1000000000.0f

// ---------------- WMMA types ----------------
typedef __attribute__((ext_vector_type(16))) __bf16 v16bf;
typedef __attribute__((ext_vector_type(8)))  float  v8f;

union V16 { v16bf v; unsigned short u[16]; };

// ---------------- helpers ----------------
static __device__ __forceinline__ unsigned short f2bf(float f) {
  unsigned u = __float_as_uint(f);
  u += 0x7FFFu + ((u >> 16) & 1u);        // round-to-nearest-even
  return (unsigned short)(u >> 16);
}
// order-preserving float<->u32 key (monotonic for non-NaN): enables atomicMax on floats
static __device__ __forceinline__ unsigned fkey(float f) {
  unsigned u = __float_as_uint(f);
  return ((int)u < 0) ? ~u : (u | 0x80000000u);
}
static __device__ __forceinline__ float funkey(unsigned k) {
  return __uint_as_float(((int)k < 0) ? (k & 0x7FFFFFFFu) : ~k);
}

// build 8 elements of the |r1-r2| and r1*r2 A-tiles from padded bf16 rows
static __device__ __forceinline__ void build_dp8(const unsigned short* r1row,
                                                 const unsigned short* r2row, int kb,
                                                 unsigned short* dd, unsigned short* pp) {
  uint4 a = *(const uint4*)(r1row + kb);   // 16B aligned (kb multiple of 8)
  uint4 c = *(const uint4*)(r2row + kb);
  unsigned av[4] = {a.x, a.y, a.z, a.w};
  unsigned cv[4] = {c.x, c.y, c.z, c.w};
#pragma unroll
  for (int i = 0; i < 4; ++i) {
    float f1 = __uint_as_float(av[i] << 16);
    float f2 = __uint_as_float(cv[i] << 16);
    dd[2 * i] = f2bf(fabsf(f1 - f2));
    pp[2 * i] = f2bf(f1 * f2);
    float g1 = __uint_as_float(av[i] & 0xFFFF0000u);
    float g2 = __uint_as_float(cv[i] & 0xFFFF0000u);
    dd[2 * i + 1] = f2bf(fabsf(g1 - g2));
    pp[2 * i + 1] = f2bf(g1 * g2);
  }
}

// ---------------- workspace layout (bytes, 256-aligned) ----------------
constexpr size_t OFF_WCT   = 0;                                  // conv weights, B-layout bf16
constexpr size_t SZ_WCT    = (size_t)KC * NPAD * 2;              // 614400
constexpr size_t OFF_WDT   = OFF_WCT + SZ_WCT;
constexpr size_t SZ_WDT    = (size_t)EPAD * NPAD * 2;            // 204800
constexpr size_t OFF_WPT   = OFF_WDT + SZ_WDT;
constexpr size_t OFF_BIASC = OFF_WPT + SZ_WDT;                   // padded conv_b (f32)
constexpr size_t OFF_BIAS0 = OFF_BIASC + 1536;                   // padded fc0_b (f32)
constexpr size_t OFF_RF32  = OFF_BIAS0 + 1536;                   // encoded r, f32, stride NPAD
constexpr size_t SZ_RF32   = (size_t)2 * BB * LL * NPAD * 4;
constexpr size_t OFF_RBF16 = OFF_RF32 + SZ_RF32;                 // encoded r, bf16, stride EPAD
constexpr size_t SZ_RBF16  = (size_t)2 * BB * LL * EPAD * 2;
constexpr size_t OFF_S1K   = OFF_RBF16 + SZ_RBF16;               // s1 as float-keys
constexpr size_t SZ_SK     = (size_t)BB * LL * NPAD * 4;
constexpr size_t OFF_S2K   = OFF_S1K + SZ_SK;
constexpr size_t OFF_EMBP  = OFF_S2K + SZ_SK;                    // bf16 emb, V+1 rows x EPAD
constexpr size_t SZ_EMBP   = (size_t)(VV + 1) * EPAD * 2;        // ~32 MB

// ---------------- kernel 0: pad + convert embedding table to bf16 ----------------
// Row VV is an all-zero row used for out-of-range conv taps; cols >= EE are zero.
__global__ void prep_emb(const float* __restrict__ emb, unsigned short* __restrict__ embp) {
  int idx = blockIdx.x * blockDim.x + threadIdx.x;
  const int total = (VV + 1) * EPAD;
  if (idx >= total) return;
  int row = idx / EPAD, col = idx % EPAD;
  float v = (row < VV && col < EE) ? emb[(size_t)row * EE + col] : 0.0f;
  embp[idx] = f2bf(v);
}

// ---------------- kernel 1: repack weights to WMMA-B bf16 layout ----------------
__global__ void prep_weights(const float* conv_w, const float* conv_b,
                             const float* fc0_w, const float* fc0_b,
                             unsigned short* WcT, unsigned short* WdT,
                             unsigned short* WpT, float* biasc, float* bias0) {
  int idx = blockIdx.x * blockDim.x + threadIdx.x;
  const int NC_ = KC * NPAD;
  const int ND_ = EPAD * NPAD;
  if (idx < NC_) {
    int k = idx / NPAD, n = idx % NPAD;
    int seg = k / EPAD, kk = k % EPAD;   // tap index, input channel
    float v = (kk < EE && n < EE) ? conv_w[n * (EE * 3) + kk * 3 + seg] : 0.0f;
    WcT[idx] = f2bf(v);
    return;
  }
  idx -= NC_;
  if (idx < ND_) {
    int k = idx / NPAD, n = idx % NPAD;
    bool ok = (k < EE && n < F0C);
    WdT[idx] = f2bf(ok ? fc0_w[n * (2 * EE) + k] : 0.0f);        // Wd = fc0_w[:, :E]
    WpT[idx] = f2bf(ok ? fc0_w[n * (2 * EE) + EE + k] : 0.0f);   // Wp = fc0_w[:, E:]
    return;
  }
  idx -= ND_;
  if (idx < NPAD) {
    biasc[idx] = (idx < EE)  ? conv_b[idx] : 0.0f;
    bias0[idx] = (idx < F0C) ? fc0_b[idx]  : 0.0f;
  }
}

// ---------------- kernel 2: init s-buffers to key(NEG) ----------------
__global__ void init_bufs(unsigned* s1k, unsigned* s2k) {
  int idx = blockIdx.x * blockDim.x + threadIdx.x;
  const int NS = BB * LL * NPAD;
  if (idx < NS) {
    unsigned kn = fkey(NEGV);
    s1k[idx] = kn;
    s2k[idx] = kn;
  }
}

// ---------------- kernel 3: encode = embed + conv(k=3) + tanh, as WMMA GEMM ----------------
// M = 2*B*L rows (q1 then q2), K = 960 (3 taps x 320 zero-padded), N = 320.
// Block = 320 threads = 10 waves; each wave owns exactly 2 n-tiles. All loads are
// unconditional aligned global loads (zero row handles conv boundary padding).
__global__ void encode_gemm(const int* __restrict__ q1, const int* __restrict__ q2,
                            const unsigned short* __restrict__ embp,
                            const unsigned short* __restrict__ WcT,
                            const float* __restrict__ biasc,
                            float* __restrict__ r_f32, unsigned short* __restrict__ r_bf16) {
  const int m0 = blockIdx.x * 16;
  const int w  = threadIdx.x >> 5;         // 0..9
  const int tl = threadIdx.x & 31;
  const int kh = tl >> 4;                  // A-layout k-half
  const int mr = m0 + (tl & 15);           // A-layout row
  int src = (mr >= BB * LL) ? 1 : 0;
  int rem = mr - src * BB * LL;
  int b = rem / LL, l = rem % LL;
  const int* q = src ? q2 : q1;
  const unsigned short* rowp[3];
#pragma unroll
  for (int s = 0; s < 3; ++s) {
    int pos = l + s - 1;
    int row = (pos >= 0 && pos < LL) ? q[b * LL + pos] : VV;   // VV = zero row
    rowp[s] = embp + (size_t)row * EPAD;
  }
  const int ntile[2] = {w, w + 10};        // exactly 2 tiles per wave
  v8f acc[2];
#pragma unroll
  for (int a = 0; a < 2; ++a) {
    float bv = biasc[ntile[a] * 16 + (tl & 15)];
#pragma unroll
    for (int v = 0; v < 8; ++v) acc[a][v] = bv;
  }
#pragma unroll
  for (int seg = 0; seg < 3; ++seg) {
#pragma unroll
    for (int tt = 0; tt < 10; ++tt) {
      const int k0 = (seg * 10 + tt) * 32;
      const int kk0 = tt * 32 + 8 * kh;    // offset within tap segment
      V16 A;
      *(uint4*)&A.u[0] = *(const uint4*)(rowp[seg] + kk0);
      *(uint4*)&A.u[8] = *(const uint4*)(rowp[seg] + kk0 + 16);
#pragma unroll
      for (int a = 0; a < 2; ++a) {
        V16 Bt;
#pragma unroll
        for (int s = 0; s < 16; ++s)
          Bt.u[s] = WcT[(size_t)(k0 + 8 * kh + (s & 7) + ((s >> 3) << 4)) * NPAD + ntile[a] * 16 + (tl & 15)];
        acc[a] = __builtin_amdgcn_wmma_f32_16x16x32_bf16(false, A.v, false, Bt.v,
                                                         (short)0, acc[a], false, false);
      }
    }
  }
#pragma unroll
  for (int a = 0; a < 2; ++a) {
    const int colf = ntile[a] * 16 + (tl & 15);
#pragma unroll
    for (int v = 0; v < 8; ++v) {
      int m = m0 + v + 8 * kh;             // C-layout row
      float val = tanhf(acc[a][v]);
      r_f32[(size_t)m * NPAD + colf] = val;
      r_bf16[(size_t)m * EPAD + colf] = f2bf(val);
    }
  }
}

// ---------------- kernel 4: fused joint GEMM + tanh + masked maxes ----------------
// One workgroup per (b, i-tile of 16). joint never leaves the WGP: s1/s2 held as
// running maxes (order-preserving u32 keys) in LDS, merged to global at the end.
// 96 wave-tasks = (i_local 16) x (j-tile 3) x (n-half 2); each half = constant 10 n-tiles.
__global__ void joint_gemm(const unsigned short* __restrict__ r_bf16,
                           const unsigned short* __restrict__ WdT,
                           const unsigned short* __restrict__ WpT,
                           const float* __restrict__ bias0,
                           const int* __restrict__ q1_len, const int* __restrict__ q2_len,
                           unsigned* __restrict__ s1k, unsigned* __restrict__ s2k) {
  extern __shared__ unsigned lds[];
  unsigned* s1l = lds;                 // 16 x NPAD
  unsigned* s2l = lds + 16 * NPAD;     // 48 x NPAD
  const int b  = blockIdx.x / 3;
  const int iT = blockIdx.x % 3;
  const unsigned kneg = fkey(NEGV);
  for (int i = threadIdx.x; i < (16 + 48) * NPAD; i += blockDim.x) lds[i] = kneg;
  __syncthreads();
  const int w  = threadIdx.x >> 5;
  const int tl = threadIdx.x & 31;
  const int kh = tl >> 4;
  const int q1l = q1_len[b];
  const int q2l = q2_len[b];
#pragma unroll 1
  for (int tt = 0; tt < 12; ++tt) {
    const int tid  = w + 8 * tt;         // 0..95
    const int half = tid / 48;
    const int rem  = tid % 48;
    const int il   = rem / 3;
    const int jT   = rem % 3;
    const int i    = iT * 16 + il;
    const int j0   = jT * 16;
    const int nbase = half * 10;         // 20 n-tiles total, constant 10 per half
    const unsigned short* r1row = r_bf16 + (size_t)(b * LL + i) * EPAD;
    const unsigned short* r2row = r_bf16 + (size_t)(BB * LL + b * LL + j0 + (tl & 15)) * EPAD;
    v8f acc[10];
#pragma unroll
    for (int a = 0; a < 10; ++a) {
      float bv = bias0[(nbase + a) * 16 + (tl & 15)];
#pragma unroll
      for (int v = 0; v < 8; ++v) acc[a][v] = bv;
    }
#pragma unroll 1
    for (int t = 0; t < 10; ++t) {
      const int k0 = t * 32;
      V16 Ad, Ap;
      build_dp8(r1row, r2row, k0 + 8 * kh,      &Ad.u[0], &Ap.u[0]);
      build_dp8(r1row, r2row, k0 + 8 * kh + 16, &Ad.u[8], &Ap.u[8]);
      if (t < 9) {   // pull next K-slice of the weight tiles toward the WGP
        __builtin_prefetch(WdT + (size_t)(k0 + 32 + tl) * NPAD, 0, 1);
        __builtin_prefetch(WpT + (size_t)(k0 + 32 + tl) * NPAD, 0, 1);
      }
#pragma unroll
      for (int a = 0; a < 10; ++a) {
        const int n0 = (nbase + a) * 16;
        V16 Bd, Bp;
#pragma unroll
        for (int s = 0; s < 16; ++s) {
          const size_t bi = (size_t)(k0 + 8 * kh + (s & 7) + ((s >> 3) << 4)) * NPAD + n0 + (tl & 15);
          Bd.u[s] = WdT[bi];
          Bp.u[s] = WpT[bi];
        }
        acc[a] = __builtin_amdgcn_wmma_f32_16x16x32_bf16(false, Ad.v, false, Bd.v,
                                                         (short)0, acc[a], false, false);
        acc[a] = __builtin_amdgcn_wmma_f32_16x16x32_bf16(false, Ap.v, false, Bp.v,
                                                         (short)0, acc[a], false, false);
      }
    }
    const bool iok = (i < q1l);          // m1 mask for s2
#pragma unroll
    for (int a = 0; a < 10; ++a) {
      const int colf = (nbase + a) * 16 + (tl & 15);
      unsigned m1max = kneg;
#pragma unroll
      for (int v = 0; v < 8; ++v) {
        const int j = j0 + v + 8 * kh;
        const unsigned key = fkey(tanhf(acc[a][v]));
        if (j < q2l && key > m1max) m1max = key;          // m2 mask for s1
        if (iok) atomicMax(&s2l[j * NPAD + colf], key);
      }
      atomicMax(&s1l[il * NPAD + colf], m1max);
    }
  }
  __syncthreads();
  // s1 tile is exclusively owned: plain stores. s2 merged across the 3 i-tile blocks.
  for (int idx = threadIdx.x; idx < 16 * NPAD; idx += blockDim.x) {
    int r = idx / NPAD, c = idx % NPAD;
    s1k[(size_t)(b * LL + iT * 16 + r) * NPAD + c] = s1l[idx];
  }
  for (int idx = threadIdx.x; idx < 48 * NPAD; idx += blockDim.x) {
    int r = idx / NPAD, c = idx % NPAD;
    atomicMax(&s2k[(size_t)(b * LL + r) * NPAD + c], s2l[idx]);
  }
}

// ---------------- kernel 5: attention pooling + fc1 + fc2 (one block per batch) -----------
__global__ void head_kernel(const float* __restrict__ r_f32,
                            const unsigned* __restrict__ s1k, const unsigned* __restrict__ s2k,
                            const int* __restrict__ q1_len, const int* __restrict__ q2_len,
                            const float* __restrict__ att_w, const float* __restrict__ att_b,
                            const float* __restrict__ fc1_w, const float* __restrict__ fc1_b,
                            const float* __restrict__ fc2_w, const float* __restrict__ fc2_b,
                            float* __restrict__ out) {
  __shared__ float aw[2 * LL];
  __shared__ float hbuf[2 * F0C];
  __shared__ float jbuf[F0C];
  const int b  = blockIdx.x;
  const int w  = threadIdx.x >> 5;
  const int tl = threadIdx.x & 31;
  // attention logits: dot(concat(r, s), att_w) + att_b, per (side, l)
  for (int p = w; p < 2 * LL; p += 8) {
    const int side = p / LL, l = p % LL;
    const float* rrow = r_f32 + (size_t)(side * BB * LL + b * LL + l) * NPAD;
    const unsigned* sk = (side ? s2k : s1k) + (size_t)(b * LL + l) * NPAD;
    float part = 0.0f;
    for (int c = tl; c < EE; c += 32)
      part += rrow[c] * att_w[c] + funkey(sk[c]) * att_w[EE + c];
#pragma unroll
    for (int off = 16; off > 0; off >>= 1) part += __shfl_xor(part, off, 32);
    if (tl == 0) aw[side * LL + l] = part + att_b[0];
  }
  __syncthreads();
  if (threadIdx.x < 2) {     // masked softmax (tiny)
    const int side = threadIdx.x;
    const int len = side ? q2_len[b] : q1_len[b];
    float mx = -3.4e38f;
    for (int l = 0; l < len; ++l) mx = fmaxf(mx, aw[side * LL + l]);
    float sum = 0.0f;
    for (int l = 0; l < LL; ++l) {
      float e = (l < len) ? expf(aw[side * LL + l] - mx) : 0.0f;
      aw[side * LL + l] = e;
      sum += e;
    }
    float inv = 1.0f / sum;
    for (int l = 0; l < LL; ++l) aw[side * LL + l] *= inv;
  }
  __syncthreads();
  // h = concat(h1, h2), h_side = sum_l a[l] * s_side[l, :]
  for (int fg = threadIdx.x; fg < 2 * F0C; fg += blockDim.x) {
    const int side = fg / F0C, f = fg % F0C;
    const unsigned* sk = (side ? s2k : s1k) + (size_t)(b * LL) * NPAD + f;
    float acc = 0.0f;
    for (int l = 0; l < LL; ++l) acc += aw[side * LL + l] * funkey(sk[(size_t)l * NPAD]);
    hbuf[fg] = acc;
  }
  __syncthreads();
  for (int r = threadIdx.x; r < F0C; r += blockDim.x) {
    float acc = fc1_b[r];
    const float* wr = fc1_w + (size_t)r * (2 * F0C);
    for (int c = 0; c < 2 * F0C; ++c) acc += wr[c] * hbuf[c];
    jbuf[r] = tanhf(acc);
  }
  __syncthreads();
  if (threadIdx.x < 2) {
    const int o = threadIdx.x;
    float acc = fc2_b[o];
    for (int r = 0; r < F0C; ++r) acc += fc2_w[o * F0C + r] * jbuf[r];
    out[b * 2 + o] = acc;
  }
}

// ---------------- host launcher ----------------
extern "C" void kernel_launch(void* const* d_in, const int* in_sizes, int n_in,
                              void* d_out, int out_size, void* d_ws, size_t ws_size,
                              hipStream_t stream) {
  const int*   q1     = (const int*)d_in[0];
  const int*   q2     = (const int*)d_in[1];
  const int*   q1_len = (const int*)d_in[2];
  const int*   q2_len = (const int*)d_in[3];
  const float* emb    = (const float*)d_in[4];
  const float* conv_w = (const float*)d_in[5];
  const float* conv_b = (const float*)d_in[6];
  const float* fc0_w  = (const float*)d_in[7];
  const float* fc0_b  = (const float*)d_in[8];
  const float* fc1_w  = (const float*)d_in[9];
  const float* fc1_b  = (const float*)d_in[10];
  const float* fc2_w  = (const float*)d_in[11];
  const float* fc2_b  = (const float*)d_in[12];
  const float* att_w  = (const float*)d_in[13];
  const float* att_b  = (const float*)d_in[14];
  float* out = (float*)d_out;
  char* ws = (char*)d_ws;

  unsigned short* WcT   = (unsigned short*)(ws + OFF_WCT);
  unsigned short* WdT   = (unsigned short*)(ws + OFF_WDT);
  unsigned short* WpT   = (unsigned short*)(ws + OFF_WPT);
  float*          biasc = (float*)(ws + OFF_BIASC);
  float*          bias0 = (float*)(ws + OFF_BIAS0);
  float*          r_f32 = (float*)(ws + OFF_RF32);
  unsigned short* rbf   = (unsigned short*)(ws + OFF_RBF16);
  unsigned*       s1k   = (unsigned*)(ws + OFF_S1K);
  unsigned*       s2k   = (unsigned*)(ws + OFF_S2K);
  unsigned short* embp  = (unsigned short*)(ws + OFF_EMBP);

  {
    int total = (VV + 1) * EPAD;
    prep_emb<<<(total + 255) / 256, 256, 0, stream>>>(emb, embp);
  }
  {
    int total = KC * NPAD + EPAD * NPAD + NPAD;
    prep_weights<<<(total + 255) / 256, 256, 0, stream>>>(conv_w, conv_b, fc0_w, fc0_b,
                                                          WcT, WdT, WpT, biasc, bias0);
  }
  {
    int total = BB * LL * NPAD;
    init_bufs<<<(total + 255) / 256, 256, 0, stream>>>(s1k, s2k);
  }
  encode_gemm<<<(2 * BB * LL) / 16, 320, 0, stream>>>(q1, q2, embp, WcT, biasc, r_f32, rbf);
  joint_gemm<<<BB * 3, 256, (16 + 48) * NPAD * sizeof(unsigned), stream>>>(
      rbf, WdT, WpT, bias0, q1_len, q2_len, s1k, s2k);
  head_kernel<<<BB, 256, 0, stream>>>(r_f32, s1k, s2k, q1_len, q2_len,
                                      att_w, att_b, fc1_w, fc1_b, fc2_w, fc2_b, out);
}
